// Attention_73220602462599
// MI455X (gfx1250) — matrix operand order repaired and checked by reading, not compiled, not run on hardware
//
#include <hip/hip_runtime.h>
#include <stdint.h>

#define Bc  2
#define Tc  2048
#define Dc  1024
#define Hc  16
#define DHc 64
#define TDc 3072

typedef __attribute__((ext_vector_type(16))) __bf16          v16bf;
typedef __attribute__((ext_vector_type(8)))  float           v8f;
typedef __attribute__((ext_vector_type(16))) unsigned short  v16u;
typedef __attribute__((ext_vector_type(8)))  unsigned short  v8u;
typedef __attribute__((ext_vector_type(4)))  unsigned short  v4u;
typedef __attribute__((ext_vector_type(4)))  unsigned int    v4ui;
typedef __attribute__((ext_vector_type(8)))  int             v8i;
typedef __attribute__((ext_vector_type(4)))  int             v4i;

#if defined(__has_builtin)
#  if __has_builtin(__builtin_amdgcn_tensor_load_to_lds)
#    define HAVE_TDM 1
#  endif
#endif
#ifndef HAVE_TDM
#  define HAVE_TDM 0
#endif

__device__ __forceinline__ unsigned short f2bf(float f) {
  unsigned int u = __builtin_bit_cast(unsigned int, f);
  u += 0x7fffu + ((u >> 16) & 1u);          // round-to-nearest-even
  return (unsigned short)(u >> 16);
}

__device__ __forceinline__ v8f wmma_bf16(v16u a, v16u b, v8f c) {
  return __builtin_amdgcn_wmma_f32_16x16x32_bf16(
      false, __builtin_bit_cast(v16bf, a),
      false, __builtin_bit_cast(v16bf, b),
      (short)0, c, false, false);
}

__device__ __forceinline__ v16u cat8(v8u a, v8u b) {
  return __builtin_shufflevector(a, b, 0,1,2,3,4,5,6,7,8,9,10,11,12,13,14,15);
}

#if HAVE_TDM
// 2-D TDM tile load: bf16 tile [tile_d1][tile_d0] from row-major tensor
// (stride d0_stride elems) into LDS at ldsAddr, with row padding codes.
__device__ __forceinline__ void tdm_load_2d(unsigned ldsAddr, const void* gptr,
                                            unsigned tensor_d0, unsigned tensor_d1,
                                            unsigned tile_d0, unsigned tile_d1,
                                            unsigned d0_stride,
                                            unsigned padi, unsigned pada) {
  unsigned long long ga = (unsigned long long)(uintptr_t)gptr;
  v4ui g0 = { 1u,                                   // count=1, user descriptor
              ldsAddr,                              // lds_addr [63:32]
              (unsigned)ga,                         // global_addr [95:64]
              (unsigned)((ga >> 32) & 0x01FFFFFFu)  // global_addr [120:96]
                  | 0x80000000u };                  // type=2 ("image") [127:126]
  v8i g1 = { (int)((1u << 16)                       // data_size = 2 bytes
                   | (1u << 20)                     // pad_enable
                   | (padi << 22) | (pada << 25)),  // pad_interval / pad_amount
             (int)((tensor_d0 & 0xFFFFu) << 16),
             (int)((tensor_d0 >> 16) | ((tensor_d1 & 0xFFFFu) << 16)),
             (int)((tensor_d1 >> 16) | (tile_d0 << 16)),
             (int)tile_d1,
             (int)d0_stride, 0, 0 };
  v4i z4 = { 0, 0, 0, 0 };
#if __clang_major__ >= 23
  v8i z8 = { 0, 0, 0, 0, 0, 0, 0, 0 };
  __builtin_amdgcn_tensor_load_to_lds(g0, g1, z4, z4, z8, 0);
#else
  __builtin_amdgcn_tensor_load_to_lds(g0, g1, z4, z4, 0);
#endif
}
#endif

// ---------------------------------------------------------------------------
// Kernel 1: QKV = x @ w_qkv, scatter bf16 Q(scaled)/K/V [b][h][t][dh]
// ---------------------------------------------------------------------------
__global__ __launch_bounds__(256)
void qkv_gemm_kernel(const float* __restrict__ x, const float* __restrict__ w,
                     unsigned short* __restrict__ Qb,
                     unsigned short* __restrict__ Kb,
                     unsigned short* __restrict__ Vb) {
  const int LDA = 40, LDBT = 48;
  __shared__ __align__(64) unsigned short As[128 * 40];
  __shared__ __align__(64) unsigned short BsT[64 * 48];

  const int tid = threadIdx.x;
  const int lane = tid & 31, wid = tid >> 5;
  const int lane16 = lane & 15, hi = lane >> 4;
  const int rowBase = blockIdx.y * 128;
  const int n0 = blockIdx.x * 64;

  v8f acc[4] = {};

  for (int k0 = 0; k0 < Dc; k0 += 32) {
    if (k0 + 32 < Dc) {
      __builtin_prefetch(&x[(size_t)(rowBase + (tid & 127)) * Dc + k0 + 32], 0, 1);
      __builtin_prefetch(&w[(size_t)(k0 + 32 + (tid & 31)) * TDc + n0], 0, 1);
    }
#pragma unroll
    for (int i = tid; i < 128 * 8; i += 256) {        // A: float4 loads -> v4u stores
      int r = i >> 3, c4 = (i & 7) * 4;
      const float4 f = *(const float4*)&x[(size_t)(rowBase + r) * Dc + k0 + c4];
      v4u hv; hv[0] = f2bf(f.x); hv[1] = f2bf(f.y); hv[2] = f2bf(f.z); hv[3] = f2bf(f.w);
      *(v4u*)&As[r * LDA + c4] = hv;
    }
#pragma unroll
    for (int i = tid; i < 32 * 16; i += 256) {        // B: float4 loads, transposed store
      int r = i >> 4, c4 = (i & 15) * 4;
      const float4 f = *(const float4*)&w[(size_t)(k0 + r) * TDc + n0 + c4];
      BsT[(c4 + 0) * LDBT + r] = f2bf(f.x);
      BsT[(c4 + 1) * LDBT + r] = f2bf(f.y);
      BsT[(c4 + 2) * LDBT + r] = f2bf(f.z);
      BsT[(c4 + 3) * LDBT + r] = f2bf(f.w);
    }
    __syncthreads();

    const int m = wid * 16 + lane16;                  // A frag: 2 x ds_load_b128
    v16u a = cat8(*(const v8u*)&As[m * LDA + hi * 8],
                  *(const v8u*)&As[m * LDA + 16 + hi * 8]);
    for (int nt = 0; nt < 4; ++nt) {                  // B frag: 2 x ds_load_b128
      int col = nt * 16 + lane16;
      v16u b = cat8(*(const v8u*)&BsT[col * LDBT + hi * 8],
                    *(const v8u*)&BsT[col * LDBT + 16 + hi * 8]);
      acc[nt] = wmma_bf16(a, b, acc[nt]);
    }
    __syncthreads();
  }

  const float qscale = 0.125f;                        // DH^-0.5
  for (int nt = 0; nt < 4; ++nt)
    for (int r = 0; r < 8; ++r) {                     // C/D: M = r + 8*hi, N = lane16
      int grow = rowBase + wid * 16 + r + hi * 8;
      int bb = grow >> 11, t = grow & (Tc - 1);
      int c = n0 + nt * 16 + lane16;
      int which = c >> 10, d = c & (Dc - 1);
      int h = d >> 6, dh = d & 63;
      size_t off = (((size_t)bb * Hc + h) * Tc + t) * DHc + dh;
      float val = acc[nt][r];
      if (which == 0)      Qb[off] = f2bf(val * qscale);
      else if (which == 1) Kb[off] = f2bf(val);
      else                 Vb[off] = f2bf(val);
    }
}

// ---------------------------------------------------------------------------
// Kernel 2: causal flash attention per (b, h, 128-query tile)
// K tile staged by the Tensor Data Mover (padded to LDK=80); V transposed
// ---------------------------------------------------------------------------
__global__ __launch_bounds__(256)
void flash_attn_kernel(const unsigned short* __restrict__ Qb,
                       const unsigned short* __restrict__ Kb,
                       const unsigned short* __restrict__ Vb,
                       unsigned short* __restrict__ Ob) {
  const int LDK = 80, LDVT = 48, LDP = 40;
  __shared__ __align__(64) unsigned short Ks[32 * 80];
  __shared__ __align__(64) unsigned short VsT[64 * 48];
  __shared__ __align__(64) unsigned short Ps[8 * 16 * 40];

  const int tid = threadIdx.x;
  const int lane = tid & 31, wid = tid >> 5;
  const int lane16 = lane & 15, hi = lane >> 4;
  const int qt = blockIdx.x, h = blockIdx.y, bb = blockIdx.z;
  const int q0 = qt * 128 + wid * 16;
  const size_t headOff = (((size_t)bb * Hc + h) * Tc) * DHc;

  v16u qf[2];                                         // Q A-frags: 4 x global_load_b128
  {
    const unsigned short* qptr = Qb + headOff + (size_t)(q0 + lane16) * DHc;
    for (int c = 0; c < 2; ++c)
      qf[c] = cat8(*(const v8u*)(qptr + c * 32 + hi * 8),
                   *(const v8u*)(qptr + c * 32 + 16 + hi * 8));
  }

  v8f o[4] = {};
  float mrow[8], lrow[8];
  for (int r = 0; r < 8; ++r) { mrow[r] = -3.0e38f; lrow[r] = 0.0f; }

  const int nj = qt * 4 + 4;                          // causal early exit
  unsigned short* Pw = Ps + wid * 16 * LDP;

  for (int j = 0; j < nj; ++j) {
#if HAVE_TDM
    if (wid == 0) {                                   // TDM: K tile 32x64 bf16,
      tdm_load_2d((unsigned)(uintptr_t)&Ks[0],        // pad 32dw rows by 8dw -> LDK=80
                  Kb + headOff + (size_t)j * 32 * DHc,
                  DHc, Tc, DHc, 32, DHc, /*padi=*/4, /*pada=*/7);
      __builtin_amdgcn_s_wait_tensorcnt(0);
    }
#endif
#pragma unroll
    for (int i = tid; i < 32 * 16; i += 256) {        // V staged transposed (+K fallback)
      int r = i >> 4, c4 = (i & 15) * 4;
      size_t g = headOff + (size_t)(j * 32 + r) * DHc + c4;
#if !HAVE_TDM
      *(v4u*)&Ks[r * LDK + c4] = *(const v4u*)&Kb[g];
#endif
      v4u vv = *(const v4u*)&Vb[g];
      VsT[(c4 + 0) * LDVT + r] = vv[0];
      VsT[(c4 + 1) * LDVT + r] = vv[1];
      VsT[(c4 + 2) * LDVT + r] = vv[2];
      VsT[(c4 + 3) * LDVT + r] = vv[3];
    }
    __syncthreads();

    if (j * 32 <= q0 + 15) {                          // wave-uniform: EXEC all-ones
      v8f s[2] = {};
      for (int g = 0; g < 2; ++g) {                   // S = Q @ K^T (4 WMMAs)
        int key = g * 16 + lane16;
        for (int c = 0; c < 2; ++c) {
          v16u kb = cat8(*(const v8u*)&Ks[key * LDK + c * 32 + hi * 8],
                         *(const v8u*)&Ks[key * LDK + c * 32 + 16 + hi * 8]);
          s[g] = wmma_bf16(qf[c], kb, s[g]);
        }
      }
      for (int g = 0; g < 2; ++g) {                   // causal mask
        int kidx = j * 32 + g * 16 + lane16;
        for (int r = 0; r < 8; ++r)
          if (kidx > q0 + r + hi * 8) s[g][r] = -3.0e38f;
      }
      for (int r = 0; r < 8; ++r) {                   // online softmax (half-wave reduce)
        float mx = fmaxf(s[0][r], s[1][r]);
        mx = fmaxf(mx, __shfl_xor(mx, 1));
        mx = fmaxf(mx, __shfl_xor(mx, 2));
        mx = fmaxf(mx, __shfl_xor(mx, 4));
        mx = fmaxf(mx, __shfl_xor(mx, 8));
        float mnew = fmaxf(mrow[r], mx);
        float corr = __expf(mrow[r] - mnew);
        float p0 = __expf(s[0][r] - mnew);
        float p1 = __expf(s[1][r] - mnew);
        float rs = p0 + p1;
        rs += __shfl_xor(rs, 1);
        rs += __shfl_xor(rs, 2);
        rs += __shfl_xor(rs, 4);
        rs += __shfl_xor(rs, 8);
        lrow[r] = lrow[r] * corr + rs;
        mrow[r] = mnew;
        for (int nt = 0; nt < 4; ++nt) o[nt][r] *= corr;
        s[0][r] = p0; s[1][r] = p1;
      }

      for (int g = 0; g < 2; ++g)                     // P: D-layout -> LDS (wave transpose)
        for (int r = 0; r < 8; ++r)
          Pw[(r + hi * 8) * LDP + g * 16 + lane16] = f2bf(s[g][r]);
      __asm volatile("s_wait_dscnt 0x0" ::: "memory");

      v16u pf = cat8(*(const v8u*)&Pw[lane16 * LDP + hi * 8],
                     *(const v8u*)&Pw[lane16 * LDP + 16 + hi * 8]);
      for (int nt = 0; nt < 4; ++nt) {                // O += P @ V (4 WMMAs)
        int col = nt * 16 + lane16;
        v16u vb = cat8(*(const v8u*)&VsT[col * LDVT + hi * 8],
                       *(const v8u*)&VsT[col * LDVT + 16 + hi * 8]);
        o[nt] = wmma_bf16(pf, vb, o[nt]);
      }
    }
    __syncthreads();
  }

  for (int r = 0; r < 8; ++r) {                       // normalize + store bf16 [b][t][d]
    float inv = (lrow[r] > 0.0f) ? (1.0f / lrow[r]) : 0.0f;
    int t = q0 + r + hi * 8;
    for (int nt = 0; nt < 4; ++nt) {
      int dh = nt * 16 + lane16;
      Ob[((size_t)bb * Tc + t) * Dc + h * DHc + dh] = f2bf(o[nt][r] * inv);
    }
  }
}

// ---------------------------------------------------------------------------
// Kernel 3: out = attn @ w_o  (A tile via TDM; B f32->bf16; fp32 store)
// ---------------------------------------------------------------------------
__global__ __launch_bounds__(256)
void out_proj_kernel(const unsigned short* __restrict__ Ab,
                     const float* __restrict__ wo,
                     float* __restrict__ out) {
  const int LDA = 40, LDBT = 48;
  __shared__ __align__(64) unsigned short As[128 * 40];
  __shared__ __align__(64) unsigned short BsT[64 * 48];

  const int tid = threadIdx.x;
  const int lane = tid & 31, wid = tid >> 5;
  const int lane16 = lane & 15, hi = lane >> 4;
  const int rowBase = blockIdx.y * 128;
  const int n0 = blockIdx.x * 64;

  v8f acc[4] = {};
  for (int k0 = 0; k0 < Dc; k0 += 32) {
    if (k0 + 32 < Dc) {
      __builtin_prefetch(&Ab[(size_t)(rowBase + (tid & 127)) * Dc + k0 + 32], 0, 1);
      __builtin_prefetch(&wo[(size_t)(k0 + 32 + (tid & 31)) * Dc + n0], 0, 1);
    }
#if HAVE_TDM
    if (wid == 0) {                                   // TDM: A tile 128x32 bf16,
      tdm_load_2d((unsigned)(uintptr_t)&As[0],        // pad 16dw rows by 4dw -> LDA=40
                  Ab + (size_t)rowBase * Dc + k0,
                  Dc, (unsigned)(Bc * Tc), 32, 128, Dc, /*padi=*/3, /*pada=*/3);
      __builtin_amdgcn_s_wait_tensorcnt(0);
    }
#else
#pragma unroll
    for (int i = tid; i < 128 * 8; i += 256) {        // A: v4u copy (already bf16)
      int r = i >> 3, c4 = (i & 7) * 4;
      *(v4u*)&As[r * LDA + c4] = *(const v4u*)&Ab[(size_t)(rowBase + r) * Dc + k0 + c4];
    }
#endif
#pragma unroll
    for (int i = tid; i < 32 * 16; i += 256) {        // B: float4 loads, transposed store
      int r = i >> 4, c4 = (i & 15) * 4;
      const float4 f = *(const float4*)&wo[(size_t)(k0 + r) * Dc + n0 + c4];
      BsT[(c4 + 0) * LDBT + r] = f2bf(f.x);
      BsT[(c4 + 1) * LDBT + r] = f2bf(f.y);
      BsT[(c4 + 2) * LDBT + r] = f2bf(f.z);
      BsT[(c4 + 3) * LDBT + r] = f2bf(f.w);
    }
    __syncthreads();

    const int m = wid * 16 + lane16;
    v16u a = cat8(*(const v8u*)&As[m * LDA + hi * 8],
                  *(const v8u*)&As[m * LDA + 16 + hi * 8]);
    for (int nt = 0; nt < 4; ++nt) {
      int col = nt * 16 + lane16;
      v16u b = cat8(*(const v8u*)&BsT[col * LDBT + hi * 8],
                    *(const v8u*)&BsT[col * LDBT + 16 + hi * 8]);
      acc[nt] = wmma_bf16(a, b, acc[nt]);
    }
    __syncthreads();
  }
  for (int nt = 0; nt < 4; ++nt)
    for (int r = 0; r < 8; ++r) {
      int grow = rowBase + wid * 16 + r + hi * 8;
      int col = n0 + nt * 16 + lane16;
      out[(size_t)grow * Dc + col] = acc[nt][r];
    }
}

// ---------------------------------------------------------------------------
extern "C" void kernel_launch(void* const* d_in, const int* in_sizes, int n_in,
                              void* d_out, int out_size, void* d_ws, size_t ws_size,
                              hipStream_t stream) {
  const float* x     = (const float*)d_in[0];
  const float* w_qkv = (const float*)d_in[1];
  const float* w_o   = (const float*)d_in[2];
  float* out = (float*)d_out;

  const size_t n = (size_t)Bc * Tc * Dc;
  unsigned short* Qb = (unsigned short*)d_ws;         // bf16 ws: Q,K,V,attn = 32 MB
  unsigned short* Kb = Qb + n;
  unsigned short* Vb = Kb + n;
  unsigned short* Ab = Vb + n;

  dim3 blk(256);
  qkv_gemm_kernel<<<dim3(TDc / 64, (Bc * Tc) / 128), blk, 0, stream>>>(x, w_qkv, Qb, Kb, Vb);
  flash_attn_kernel<<<dim3(Tc / 128, Hc, Bc), blk, 0, stream>>>(Qb, Kb, Vb, Ab);
  out_proj_kernel<<<dim3(Dc / 64, (Bc * Tc) / 128), blk, 0, stream>>>(Ab, w_o, out);
}
